// MultiHeadAttention_17970143167014
// MI455X (gfx1250) — compile-verified, matrix operand order checked by reading, not run on hardware
//
#include <hip/hip_runtime.h>
#include <hip/hip_bf16.h>

// Problem constants (match reference)
#define B_   4
#define T_   2048
#define D_   1024
#define H_   16
#define DEP_ 64

typedef __bf16 bf16;
typedef __attribute__((ext_vector_type(16))) __bf16 v16bf;
typedef __attribute__((ext_vector_type(8)))  __bf16 v8bf;
typedef __attribute__((ext_vector_type(4)))  __bf16 v4bf;
typedef __attribute__((ext_vector_type(8)))  float  v8f;
typedef __attribute__((ext_vector_type(4)))  int    v4i;

#define AS1 __attribute__((address_space(1)))
#define AS3 __attribute__((address_space(3)))

__device__ __forceinline__ v8f wmma_bf(v16bf a, v16bf b, v8f c) {
  // D = A(16x32 bf16) x B(32x16 bf16) + C(16x16 f32)
  return __builtin_amdgcn_wmma_f32_16x16x32_bf16(
      /*neg_a=*/false, a, /*neg_b=*/false, b,
      /*c_mod=*/(short)0, c, /*reuse_a=*/false, /*reuse_b=*/false);
}

// 16-byte global->LDS copy; async on gfx1250 toolchains exposing the builtin.
__device__ __forceinline__ void stage16(const bf16* gsrc, bf16* ldst) {
#if defined(__has_builtin) && __has_builtin(__builtin_amdgcn_global_load_async_to_lds_b128)
  __builtin_amdgcn_global_load_async_to_lds_b128(
      (AS1 v4i*)(uintptr_t)gsrc,
      (AS3 v4i*)(uint32_t)(uintptr_t)ldst, 0, 0);
#else
  *reinterpret_cast<v8bf*>(ldst) = *reinterpret_cast<const v8bf*>(gsrc);
#endif
}

__device__ __forceinline__ void stage_wait() {
#if defined(__has_builtin) && __has_builtin(__builtin_amdgcn_global_load_async_to_lds_b128)
  asm volatile("s_wait_asynccnt 0" ::: "memory");
#endif
}

// A/B fragment loader for a row-major bf16 matrix (16 rows x 32 "K" cols).
// Wave32 layout: lanes 0-15 -> row=lane,   K = {0..7, 16..23}
//                lanes 16-31 -> row=lane-16, K = {8..15, 24..31}
__device__ __forceinline__ v16bf load_frag16(const bf16* base, int ld, int lane) {
  const int row = lane & 15;
  const int kg  = (lane >> 4) << 3;          // 0 or 8
  const v8bf a = *reinterpret_cast<const v8bf*>(base + (size_t)row * ld + kg);
  const v8bf b = *reinterpret_cast<const v8bf*>(base + (size_t)row * ld + kg + 16);
  v16bf r;
#pragma unroll
  for (int i = 0; i < 8; ++i) { r[i] = a[i]; r[i + 8] = b[i]; }
  return r;
}

// ---------------- pre-pass: f32 -> bf16 conversions ----------------

__global__ __launch_bounds__(256) void cvt4_kernel(const float* __restrict__ x,
                                                   bf16* __restrict__ xb) {
  size_t i = ((size_t)blockIdx.x * blockDim.x + threadIdx.x) * 4;
  float4 f = *reinterpret_cast<const float4*>(x + i);
  v4bf o;
  o[0] = (bf16)f.x; o[1] = (bf16)f.y; o[2] = (bf16)f.z; o[3] = (bf16)f.w;
  *reinterpret_cast<v4bf*>(xb + i) = o;
}

// w: K x N row-major (f32)  ->  wt: N x K row-major (bf16)  == W^T
__global__ __launch_bounds__(256) void transpose_cvt_kernel(const float* __restrict__ w,
                                                            bf16* __restrict__ wt,
                                                            int K, int N) {
  int idx = blockIdx.x * blockDim.x + threadIdx.x;
  if (idx >= K * N) return;
  int n = idx / K;
  int k = idx % K;
  wt[(size_t)n * K + k] = (bf16)w[(size_t)k * N + n];
}

// ---------------- kernel 1: fused QKV projection ----------------
// Block = 8 waves sharing one 16-row A stripe (staged async into LDS),
// each wave owns one 16-col N tile. Epilogue scatters Q/K/V.

__global__ __launch_bounds__(256) void qkv_gemm_kernel(
    const bf16* __restrict__ xb, const bf16* __restrict__ waT,
    const float* __restrict__ b_attn,
    bf16* __restrict__ Qb, bf16* __restrict__ Kb, bf16* __restrict__ Vt,
    float* __restrict__ present) {
  __shared__ __align__(16) bf16 Ash[16 * D_];      // 32 KB A stripe
  const int lane = threadIdx.x & 31;
  const int wave = threadIdx.x >> 5;
  const int NGRP = (3 * D_ / 16) / 8;              // 24 groups of 8 N-tiles
  int mt = blockIdx.x / NGRP;
  int nt = (blockIdx.x % NGRP) * 8 + wave;
  int m0 = mt * 16, n0 = nt * 16;

  // cooperative async stage of A tile (16 x 1024 bf16)
  {
    const bf16* src = xb + (size_t)m0 * D_;
#pragma unroll
    for (int c = threadIdx.x; c < (16 * D_) / 8; c += 256)
      stage16(src + (size_t)c * 8, &Ash[c * 8]);
    stage_wait();
  }
  __syncthreads();

  const bf16* Brow = waT + (size_t)n0 * D_;
  v8f acc = {};
#pragma unroll 4
  for (int k = 0; k < D_; k += 32) {
    v16bf af = load_frag16(&Ash[k], D_, lane);       // ds_load_b128
    v16bf bf_ = load_frag16(Brow + k, D_, lane);     // global_load_b128
    acc = wmma_bf(af, bf_, acc);
  }

  // C layout: VGPR i: lanes 0-15 -> M=i, lanes 16-31 -> M=8+i ; N=lane&15
  const int ml = (lane >> 4) << 3;
  const int nl = lane & 15;
  const int n  = n0 + nl;
  const float bias = b_attn[n];
  const int seg = n0 / D_;                          // 0=Q 1=K 2=V (tile-uniform)
  const int nn  = n - seg * D_;
  const int h = nn >> 6, d = nn & 63;

#pragma unroll
  for (int i = 0; i < 8; ++i) {
    int m = m0 + ml + i;                            // global token index
    int b = m / T_, t = m % T_;
    float val = acc[i] + bias;
    size_t bh = (size_t)b * H_ + h;
    if (seg == 0) {
      Qb[(bh * T_ + t) * DEP_ + d] = (bf16)val;
    } else if (seg == 1) {
      present[((((size_t)b * 2 + 0) * H_ + h) * T_ + t) * DEP_ + d] = val;
      Kb[(bh * T_ + t) * DEP_ + d] = (bf16)val;
    } else {
      present[((((size_t)b * 2 + 1) * H_ + h) * T_ + t) * DEP_ + d] = val;
      Vt[(bh * DEP_ + d) * (size_t)T_ + t] = (bf16)val;  // transposed V
    }
  }
}

// ---------------- kernel 2: causal flash attention ----------------
// Block = 8 waves covering a 128-query band of one (b,h). Each 32-key
// K/V tile is staged once per block into LDS (8x global-traffic cut).

__global__ __launch_bounds__(256) void attn_kernel(
    const bf16* __restrict__ Qb, const bf16* __restrict__ Kb,
    const bf16* __restrict__ Vt, bf16* __restrict__ attnb) {
  __shared__ __align__(16) bf16 Ksh[32 * DEP_];     // 4 KB: keys x depth
  __shared__ __align__(16) bf16 Vsh[DEP_ * 32];     // 4 KB: depth x keys
  __shared__ __align__(16) bf16 Plds[8][16 * 32];   // 8 KB: per-wave P tile

  const int lane = threadIdx.x & 31;
  const int wave = threadIdx.x >> 5;
  const int QB = T_ / 128;                           // 16 query bands
  int bh = blockIdx.x / QB;                          // b*H + h
  int qb = blockIdx.x % QB;
  int q0 = qb * 128 + wave * 16;

  const bf16* Qw = Qb + (size_t)bh * T_ * DEP_;
  const bf16* Kw = Kb + (size_t)bh * T_ * DEP_;
  const bf16* Vw = Vt + (size_t)bh * DEP_ * T_;

  v16bf qf0 = load_frag16(Qw + (size_t)q0 * DEP_ + 0,  DEP_, lane);
  v16bf qf1 = load_frag16(Qw + (size_t)q0 * DEP_ + 32, DEP_, lane);

  v8f o0 = {}, o1 = {}, o2 = {}, o3 = {};
  float mrow[8], lrow[8];
#pragma unroll
  for (int i = 0; i < 8; ++i) { mrow[i] = -3.0e38f; lrow[i] = 0.0f; }

  const int ml = (lane >> 4) << 3;
  const int nl = lane & 15;
  const int nblk = qb * 4 + 4;                       // block-uniform trip count
  const float scale = 0.125f;                        // 1/sqrt(64)

  for (int j = 0; j < nblk; ++j) {
    int k0 = j * 32;
    // stage K (32x64) and V^T (64x32): 512 x 16B chunks over 256 threads
    {
      int c = threadIdx.x;
      int kr = c >> 3, kc = (c & 7) * 8;             // K: 256 chunks
      stage16(Kw + (size_t)(k0 + kr) * DEP_ + kc, &Ksh[kr * DEP_ + kc]);
      int vr = c >> 2, vc = (c & 3) * 8;             // V: 256 chunks
      stage16(Vw + (size_t)vr * T_ + k0 + vc, &Vsh[vr * 32 + vc]);
      stage_wait();
    }
    __syncthreads();

    v16bf kf00 = load_frag16(&Ksh[0],              DEP_, lane);
    v16bf kf01 = load_frag16(&Ksh[32],             DEP_, lane);
    v16bf kf10 = load_frag16(&Ksh[16 * DEP_],      DEP_, lane);
    v16bf kf11 = load_frag16(&Ksh[16 * DEP_ + 32], DEP_, lane);
    v8f z = {};
    v8f s0 = wmma_bf(qf1, kf01, wmma_bf(qf0, kf00, z));
    v8f s1 = wmma_bf(qf1, kf11, wmma_bf(qf0, kf10, z));

#pragma unroll
    for (int i = 0; i < 8; ++i) {
      int qg = q0 + ml + i;
      // additive causal mask exactly as reference: masked -> -1e10
      float a0 = (k0 + nl      <= qg) ? s0[i] * scale : -1.0e10f;
      float a1 = (k0 + 16 + nl <= qg) ? s1[i] * scale : -1.0e10f;
      // row-max across the 16 lanes holding this row
      float rm = fmaxf(a0, a1);
      rm = fmaxf(rm, __shfl_xor(rm, 1, 32));
      rm = fmaxf(rm, __shfl_xor(rm, 2, 32));
      rm = fmaxf(rm, __shfl_xor(rm, 4, 32));
      rm = fmaxf(rm, __shfl_xor(rm, 8, 32));
      float mnew = fmaxf(mrow[i], rm);
      float corr = __expf(mrow[i] - mnew);
      float p0 = __expf(a0 - mnew);
      float p1 = __expf(a1 - mnew);
      float rs = p0 + p1;
      rs += __shfl_xor(rs, 1, 32);
      rs += __shfl_xor(rs, 2, 32);
      rs += __shfl_xor(rs, 4, 32);
      rs += __shfl_xor(rs, 8, 32);
      lrow[i] = lrow[i] * corr + rs;
      mrow[i] = mnew;
      o0[i] *= corr; o1[i] *= corr; o2[i] *= corr; o3[i] *= corr;
      // stash P (C layout) row-major into LDS for A-fragment re-read
      Plds[wave][(ml + i) * 32 + nl]      = (bf16)p0;
      Plds[wave][(ml + i) * 32 + 16 + nl] = (bf16)p1;
    }
    asm volatile("s_wait_dscnt 0" ::: "memory");
    v16bf pf  = load_frag16(&Plds[wave][0], 32, lane);   // A: 16x32 probs
    v16bf vf0 = load_frag16(&Vsh[ 0 * 32], 32, lane);
    v16bf vf1 = load_frag16(&Vsh[16 * 32], 32, lane);
    v16bf vf2 = load_frag16(&Vsh[32 * 32], 32, lane);
    v16bf vf3 = load_frag16(&Vsh[48 * 32], 32, lane);
    o0 = wmma_bf(pf, vf0, o0);
    o1 = wmma_bf(pf, vf1, o1);
    o2 = wmma_bf(pf, vf2, o2);
    o3 = wmma_bf(pf, vf3, o3);
    __syncthreads();                                  // protect Ksh/Vsh reuse
  }

  int b = bh / H_, h = bh % H_;
#pragma unroll
  for (int i = 0; i < 8; ++i) {
    float rl = 1.0f / lrow[i];
    int q = q0 + ml + i;
    size_t rowoff = ((size_t)b * T_ + q) * D_ + (size_t)h * DEP_;
    attnb[rowoff +  0 + nl] = (bf16)(o0[i] * rl);
    attnb[rowoff + 16 + nl] = (bf16)(o1[i] * rl);
    attnb[rowoff + 32 + nl] = (bf16)(o2[i] * rl);
    attnb[rowoff + 48 + nl] = (bf16)(o3[i] * rl);
  }
}

// ---------------- kernel 3: output projection ----------------

__global__ __launch_bounds__(256) void proj_gemm_kernel(
    const bf16* __restrict__ attnb, const bf16* __restrict__ wpT,
    const float* __restrict__ b_proj, float* __restrict__ out) {
  __shared__ __align__(16) bf16 Ash[16 * D_];      // 32 KB A stripe
  const int lane = threadIdx.x & 31;
  const int wave = threadIdx.x >> 5;
  const int NGRP = (D_ / 16) / 8;                  // 8 groups of 8 N-tiles
  int mt = blockIdx.x / NGRP;
  int nt = (blockIdx.x % NGRP) * 8 + wave;
  int m0 = mt * 16, n0 = nt * 16;

  {
    const bf16* src = attnb + (size_t)m0 * D_;
#pragma unroll
    for (int c = threadIdx.x; c < (16 * D_) / 8; c += 256)
      stage16(src + (size_t)c * 8, &Ash[c * 8]);
    stage_wait();
  }
  __syncthreads();

  const bf16* Brow = wpT + (size_t)n0 * D_;
  v8f acc = {};
#pragma unroll 4
  for (int k = 0; k < D_; k += 32) {
    acc = wmma_bf(load_frag16(&Ash[k], D_, lane),
                  load_frag16(Brow + k, D_, lane), acc);
  }
  const int ml = (lane >> 4) << 3;
  const int nl = lane & 15;
  const int n = n0 + nl;
  const float bias = b_proj[n];
#pragma unroll
  for (int i = 0; i < 8; ++i) {
    int m = m0 + ml + i;
    out[(size_t)m * D_ + n] = acc[i] + bias;
  }
}

// ---------------- host launcher ----------------

extern "C" void kernel_launch(void* const* d_in, const int* in_sizes, int n_in,
                              void* d_out, int out_size, void* d_ws, size_t ws_size,
                              hipStream_t stream) {
  (void)in_sizes; (void)n_in; (void)out_size; (void)ws_size;
  const float* x      = (const float*)d_in[0];
  const float* w_attn = (const float*)d_in[1];
  const float* b_attn = (const float*)d_in[2];
  const float* w_proj = (const float*)d_in[3];
  const float* b_proj = (const float*)d_in[4];

  float* out     = (float*)d_out;                       // (B,T,D)
  float* present = out + (size_t)B_ * T_ * D_;          // (B,2,H,T,64)

  const size_t nBTD = (size_t)B_ * T_ * D_;             // 8388608
  bf16* ws    = (bf16*)d_ws;
  bf16* xb    = ws;  ws += nBTD;                        // x in bf16
  bf16* waT   = ws;  ws += (size_t)3 * D_ * D_;         // w_attn^T bf16
  bf16* wpT   = ws;  ws += (size_t)D_ * D_;             // w_proj^T bf16
  bf16* Qb    = ws;  ws += nBTD;                        // (B,H,T,64)
  bf16* Kb    = ws;  ws += nBTD;                        // (B,H,T,64)
  bf16* Vt    = ws;  ws += nBTD;                        // (B,H,64,T)
  bf16* attnb = ws;                                     // (B,T,D)

  // pre-pass conversions
  cvt4_kernel<<<(int)(nBTD / (256 * 4)), 256, 0, stream>>>(x, xb);
  transpose_cvt_kernel<<<(3 * D_ * D_ + 255) / 256, 256, 0, stream>>>(
      w_attn, waT, D_, 3 * D_);
  transpose_cvt_kernel<<<(D_ * D_ + 255) / 256, 256, 0, stream>>>(
      w_proj, wpT, D_, D_);

  // QKV projection: (8192/16) M-tiles x 24 N-groups = 12288 blocks
  qkv_gemm_kernel<<<(B_ * T_ / 16) * ((3 * D_ / 16) / 8), 256, 0, stream>>>(
      xb, waT, b_attn, Qb, Kb, Vt, present);

  // attention: B*H * (T/128) query bands = 1024 blocks
  attn_kernel<<<B_ * H_ * (T_ / 128), 256, 0, stream>>>(Qb, Kb, Vt, attnb);

  // output projection: (8192/16) M-tiles x 8 N-groups = 4096 blocks
  proj_gemm_kernel<<<(B_ * T_ / 16) * ((D_ / 16) / 8), 256, 0, stream>>>(
      attnb, wpT, b_proj, out);
}
